// GCNEncoder_3178275799751
// MI455X (gfx1250) — compile-verified, matrix operand order checked by reading, not run on hardware
//
#include <hip/hip_runtime.h>
#include <hip/hip_bf16.h>
#include <stdint.h>

typedef __attribute__((ext_vector_type(16))) __bf16 v16bf;
typedef __attribute__((ext_vector_type(8)))  __bf16 v8bf;
typedef __attribute__((ext_vector_type(8)))  float  v8f;

// ---------- helpers ----------
__device__ __forceinline__ __bf16 f2bf(float f) {
    union { float f; unsigned u; } uf; uf.f = f;
    unsigned u = uf.u;
    u += 0x7fffu + ((u >> 16) & 1u);          // round-to-nearest-even
    unsigned short h = (unsigned short)(u >> 16);
    __bf16 r; __builtin_memcpy(&r, &h, 2);
    return r;
}

// Load one 16x32 bf16 WMMA fragment (A-layout; B uses same layout on the
// N-major transposed weight matrix). Per ISA 7.12.2:
//   lane L: row = L&15, K-groups = {(L>>4)*8 .. +7} and {+16 .. +23}
__device__ __forceinline__ v16bf load_frag(const __bf16* __restrict__ base,
                                           int ld, int tileRow, int k0, int lane) {
    int r  = tileRow + (lane & 15);
    int kg = (lane >> 4) * 8;
    const __bf16* p = base + (size_t)r * ld + k0 + kg;
    v8bf lo = *(const v8bf*)p;          // K = kg .. kg+7       (16B)
    v8bf hi = *(const v8bf*)(p + 16);   // K = kg+16 .. kg+23   (16B)
    return __builtin_shufflevector(lo, hi, 0,1,2,3,4,5,6,7,8,9,10,11,12,13,14,15);
}

// ---------- degree / norm ----------
__global__ void init_deg(float* __restrict__ deg, int n) {
    int i = blockIdx.x * blockDim.x + threadIdx.x;
    if (i < n) deg[i] = 1.0f;                       // self-loop
}
__global__ void count_deg(const long long* __restrict__ ei, float* __restrict__ deg, int E) {
    int e = blockIdx.x * blockDim.x + threadIdx.x;
    if (e < E) atomicAdd(&deg[ei[(size_t)E + e]], 1.0f);
}
__global__ void calc_dinv(const float* __restrict__ deg, float* __restrict__ dinv, int n) {
    int i = blockIdx.x * blockDim.x + threadIdx.x;
    if (i < n) dinv[i] = rsqrtf(deg[i]);
}

// ---------- fp32 -> bf16 conversions ----------
__global__ void cvt_bf16(const float* __restrict__ in, __bf16* __restrict__ out, long long n) {
    long long i = (long long)blockIdx.x * blockDim.x + threadIdx.x;
    if (i < n) out[i] = f2bf(in[i]);
}
// W is [K,N] row-major; produce WT [N,K] (so B-fragments are contiguous in K)
__global__ void transpose_cvt(const float* __restrict__ W, __bf16* __restrict__ WT, int K, int N) {
    int t = blockIdx.x * blockDim.x + threadIdx.x;
    if (t < K * N) {
        int k = t % K, n = t / K;
        WT[(size_t)n * K + k] = f2bf(W[(size_t)k * N + n]);
    }
}

// ---------- WMMA GEMM: C[M,N'] = A[M,K] * BT[N',K]^T  (bf16 in, f32 out) ----------
// 256 threads = 8 waves = 2 (M) x 4 (N); each wave does NT 16x16 tiles along N.
template <int K, int NT>
__global__ __launch_bounds__(256) void gemm_wmma(const __bf16* __restrict__ A,
                                                 const __bf16* __restrict__ BT,
                                                 float* __restrict__ C,
                                                 int M, int ldc) {
    const int lane = threadIdx.x & 31;
    const int w    = threadIdx.x >> 5;
    const int wm   = w & 1;          // 2 M-tiles per block
    const int wn   = w >> 1;         // 4 wave-columns
    const int rowBase = (blockIdx.x * 2 + wm) * 16;
    const int colBase = wn * NT * 16;
    if (rowBase >= M) return;

    v8f acc[NT] = {};
#pragma unroll
    for (int k0 = 0; k0 < K; k0 += 32) {
        v16bf a = load_frag(A, K, rowBase, k0, lane);
#pragma unroll
        for (int t = 0; t < NT; ++t) {
            v16bf b = load_frag(BT, K, colBase + t * 16, k0, lane);
            acc[t] = __builtin_amdgcn_wmma_f32_16x16x32_bf16(
                false, a, false, b, (short)0, acc[t], false, false);
        }
    }
    // C/D layout: VGPR v, lanes 0-15 -> M=v, lanes 16-31 -> M=v+8; N = lane&15
    const int r0 = rowBase + 8 * (lane >> 4);
    const int c0 = lane & 15;
#pragma unroll
    for (int t = 0; t < NT; ++t)
#pragma unroll
        for (int v = 0; v < 8; ++v)
            C[(size_t)(r0 + v) * ldc + colBase + t * 16 + c0] = acc[t][v];
}

// ---------- aggregation ----------
// acc[i,:] = h[i,:] * dinv[i]^2   (self-loop term; also zero-initializes acc)
__global__ void selfloop_init(const float* __restrict__ h, const float* __restrict__ dinv,
                              float* __restrict__ acc, long long total, int F) {
    long long t = (long long)blockIdx.x * blockDim.x + threadIdx.x;
    if (t < total) {
        long long node = t / F;
        float di = dinv[node];
        acc[t] = h[t] * di * di;
    }
}

// Edge scatter: acc[dst,:] += h[src,:] * dinv[src]*dinv[dst]
// F/FPT lanes per edge -> fully coalesced 1KB/512B row gathers.
template <int F, int FPT>
__global__ void scatter_add(const float* __restrict__ h, const long long* __restrict__ ei,
                            const float* __restrict__ dinv, float* __restrict__ acc, int E) {
    constexpr int LPE = F / FPT;
    long long tid  = (long long)blockIdx.x * blockDim.x + threadIdx.x;
    long long edge = tid / LPE;
    int       ln   = (int)(tid % LPE);
    if (edge >= E) return;
    long long s = ei[edge];
    long long d = ei[(size_t)E + edge];
    float nm = dinv[s] * dinv[d];
    const float* hp = h  + (size_t)s * F + ln * FPT;
    float*       op = acc + (size_t)d * F + ln * FPT;
    float4 v0 = *(const float4*)hp;
    float4 v1 = *(const float4*)(hp + 4);
    atomicAdd(op + 0, v0.x * nm); atomicAdd(op + 1, v0.y * nm);
    atomicAdd(op + 2, v0.z * nm); atomicAdd(op + 3, v0.w * nm);
    atomicAdd(op + 4, v1.x * nm); atomicAdd(op + 5, v1.y * nm);
    atomicAdd(op + 6, v1.z * nm); atomicAdd(op + 7, v1.w * nm);
}

// hb = bf16(relu(acc + b))
__global__ void bias_relu_cvt(const float* __restrict__ acc, const float* __restrict__ b,
                              __bf16* __restrict__ hb, long long total, int F) {
    long long t = (long long)blockIdx.x * blockDim.x + threadIdx.x;
    if (t < total) {
        float v = acc[t] + b[t % F];
        hb[t] = f2bf(v > 0.0f ? v : 0.0f);
    }
}

// out = h2 * dinv^2 + b   (self-loop term + bias; initializes d_out)
__global__ void out_init(const float* __restrict__ h2, const float* __restrict__ dinv,
                         const float* __restrict__ b, float* __restrict__ out,
                         long long total, int F) {
    long long t = (long long)blockIdx.x * blockDim.x + threadIdx.x;
    if (t < total) {
        long long node = t / F;
        float di = dinv[node];
        out[t] = h2[t] * di * di + b[t % F];
    }
}

// ---------- host ----------
static inline int cdiv(long long a, long long b) { return (int)((a + b - 1) / b); }

extern "C" void kernel_launch(void* const* d_in, const int* in_sizes, int n_in,
                              void* d_out, int out_size, void* d_ws, size_t ws_size,
                              hipStream_t stream) {
    const float*     x   = (const float*)d_in[0];
    const long long* ei  = (const long long*)d_in[1];
    const float*     W1  = (const float*)d_in[2];
    const float*     b1  = (const float*)d_in[3];
    const float*     W2  = (const float*)d_in[4];
    const float*     b2  = (const float*)d_in[5];
    float*           out = (float*)d_out;

    const int F_IN = 128, F_H = 256, F_OUT = 128;
    const int N = in_sizes[0] / F_IN;     // 100000
    const int E = in_sizes[1] / 2;        // 1600000

    // workspace carve (256B aligned), with reuse to cap peak footprint
    char* p = (char*)d_ws;
    auto carve = [&](size_t bytes) {
        char* r = p;
        p += (bytes + 255) & ~(size_t)255;
        return r;
    };
    float*  deg  = (float*)carve((size_t)N * 4);
    float*  dinv = (float*)carve((size_t)N * 4);
    __bf16* xb   = (__bf16*)carve((size_t)N * F_IN * 2);
    __bf16* w1t  = (__bf16*)carve((size_t)F_IN * F_H * 2);
    __bf16* w2t  = (__bf16*)carve((size_t)F_H * F_OUT * 2);
    char*  bigA  = carve((size_t)N * F_H * 4);   // h1 (f32) -> later hb (bf16)
    char*  bigB  = carve((size_t)N * F_H * 4);   // agg1 (f32) -> later h2 (f32)
    float*  h1   = (float*)bigA;
    __bf16* hb   = (__bf16*)bigA;
    float*  agg1 = (float*)bigB;
    float*  h2   = (float*)bigB;

    const int T = 256;

    // --- normalization ---
    init_deg <<<cdiv(N, T), T, 0, stream>>>(deg, N);
    count_deg<<<cdiv(E, T), T, 0, stream>>>(ei, deg, E);
    calc_dinv<<<cdiv(N, T), T, 0, stream>>>(deg, dinv, N);

    // --- bf16 conversions ---
    cvt_bf16     <<<cdiv((long long)N * F_IN, T), T, 0, stream>>>(x, xb, (long long)N * F_IN);
    transpose_cvt<<<cdiv(F_IN * F_H, T), T, 0, stream>>>(W1, w1t, F_IN, F_H);
    transpose_cvt<<<cdiv(F_H * F_OUT, T), T, 0, stream>>>(W2, w2t, F_H, F_OUT);

    // --- layer 1: h1 = X @ W1 (WMMA) ---
    gemm_wmma<128, 4><<<cdiv(N, 32), 256, 0, stream>>>(xb, w1t, h1, N, F_H);
    // agg1 = self-loop term, then scatter edges
    selfloop_init<<<cdiv((long long)N * F_H, T), T, 0, stream>>>(h1, dinv, agg1,
                                                                 (long long)N * F_H, F_H);
    scatter_add<256, 8><<<cdiv((long long)E * (256 / 8), T), T, 0, stream>>>(h1, ei, dinv, agg1, E);
    // hb = bf16(relu(agg1 + b1))  (overwrites h1 region — h1 dead now)
    bias_relu_cvt<<<cdiv((long long)N * F_H, T), T, 0, stream>>>(agg1, b1, hb,
                                                                 (long long)N * F_H, F_H);

    // --- layer 2: h2 = hb @ W2 (WMMA; h2 overwrites agg1 region — agg1 dead) ---
    gemm_wmma<256, 2><<<cdiv(N, 32), 256, 0, stream>>>(hb, w2t, h2, N, F_OUT);
    // out = self-loop + bias, then scatter edges into d_out
    out_init<<<cdiv((long long)N * F_OUT, T), T, 0, stream>>>(h2, dinv, b2, out,
                                                              (long long)N * F_OUT, F_OUT);
    scatter_add<128, 8><<<cdiv((long long)E * (128 / 8), T), T, 0, stream>>>(h2, ei, dinv, out, E);
}
